// CrossAttention_85736137163024
// MI455X (gfx1250) — matrix operand order repaired and checked by reading, not compiled, not run on hardware
//
#include <hip/hip_runtime.h>

// ---------------------------------------------------------------------------
// Cross-attention (two streams) for MI455X / gfx1250, f16 WMMA + f32 accum.
//   B=2, N=2048, DIM=1024, HEADS=16, DIM_HEAD=64
// Register-blocked: GEMM waves own 64x64 tiles (16 WMMA accums), attention
// waves own 32 queries (two 16-row tiles sharing K/V fragments).
// ---------------------------------------------------------------------------

namespace {
constexpr int kB     = 2;
constexpr int kN     = 2048;
constexpr int kDim   = 1024;
constexpr int kHeads = 16;
constexpr int kDh    = 64;
constexpr int kM     = kB * kN;      // 4096 rows (batch*seq)
constexpr int kNcQkv = 3 * kDim;     // 3072
// softmax scale folded into Q at f16-conversion time; exp2 domain (log2 e)
constexpr float kQScale = 0.125f * 1.4426950408889634f;
}

typedef __attribute__((ext_vector_type(8)))  _Float16 v8h;
typedef __attribute__((ext_vector_type(16))) _Float16 v16h;
typedef __attribute__((ext_vector_type(8)))  float    v8f;

// ---- WMMA helper: D = A(16x32 f16) x B(32x16 f16) + C(f32) ----------------
__device__ __forceinline__ v8f wmma_f16(v16h a, v16h b, v8f c) {
  return __builtin_amdgcn_wmma_f32_16x16x32_f16(false, a, false, b,
                                                (short)0, c, false, false);
}

// ---- A-operand fragment (16xK tile, source row-major [rows, K]) -----------
// ISA layout: lane(l&15)=row; half 0 holds K {0..7,16..23}, half 1 {8..15,24..31}
__device__ __forceinline__ v16h load_a_frag(const _Float16* __restrict__ base,
                                            int ld, int row, int k, int lane) {
  const int half = lane >> 4;
  const _Float16* p = base + (size_t)(row + (lane & 15)) * ld + k + half * 8;
  v8h lo = *(const v8h*)p;          // K = 8h .. 8h+7
  v8h hi = *(const v8h*)(p + 16);   // K = 16+8h .. 16+8h+7
  return __builtin_shufflevector(lo, hi, 0,1,2,3,4,5,6,7,8,9,10,11,12,13,14,15);
}

// ---- B-operand fragment (Kx16 tile, source "transposed" row-major [cols, K])
// ISA layout: lane(l&15)=col; half 0 holds K {0..7,16..23}, half 1 {8..15,24..31}
__device__ __forceinline__ v16h load_b_frag(const _Float16* __restrict__ base,
                                            int ld, int col, int k, int lane) {
  const int half = lane >> 4;
  const _Float16* p = base + (size_t)(col + (lane & 15)) * ld + k + half * 8;
  v8h lo = *(const v8h*)p;
  v8h hi = *(const v8h*)(p + 16);
  return __builtin_shufflevector(lo, hi, 0,1,2,3,4,5,6,7,8,9,10,11,12,13,14,15);
}

// ---------------------------------------------------------------------------
// f32 -> f16 convert
__global__ void cvt_f16(const float* __restrict__ x, _Float16* __restrict__ y,
                        int n) {
  int i = blockIdx.x * blockDim.x + threadIdx.x;
  if (i < n) y[i] = (_Float16)x[i];
}

// W[K][Nc] f32  ->  Wt[Nc][K] f16  (coalesced reads)
__global__ void transpose_cvt(const float* __restrict__ W,
                              _Float16* __restrict__ Wt, int K, int Nc) {
  int i = blockIdx.x * blockDim.x + threadIdx.x;
  if (i < K * Nc) {
    int k = i / Nc, n = i - k * Nc;
    Wt[(size_t)n * K + k] = (_Float16)W[i];
  }
}

// ---------------------------------------------------------------------------
// QKV projection: [4096,1024] x [1024,3072] -> scatter q/k/vT (f16)
// 8 waves/block, each wave computes a 64x64 tile (4x4 WMMA accumulators).
__global__ __launch_bounds__(256)
void qkv_gemm(const _Float16* __restrict__ A,    // x_f16  [4096, 1024]
              const _Float16* __restrict__ Bt,   // W_qkvT [3072, 1024]
              _Float16* __restrict__ qb,         // [B*H, N, 64] pre-scaled
              _Float16* __restrict__ kb,         // [B*H, N, 64]
              _Float16* __restrict__ vt) {       // [B*H, 64, N]
  const int lane = threadIdx.x & 31, w = threadIdx.x >> 5;
  const int m0 = blockIdx.x * 256 + (w >> 1) * 64;
  const int n0 = blockIdx.y * 128 + (w & 1) * 64;

  v8f acc[4][4] = {};
  for (int k = 0; k < kDim; k += 32) {
    v16h af[4], bf[4];
#pragma unroll
    for (int i = 0; i < 4; i++) af[i] = load_a_frag(A,  kDim, m0 + 16 * i, k, lane);
#pragma unroll
    for (int j = 0; j < 4; j++) bf[j] = load_b_frag(Bt, kDim, n0 + 16 * j, k, lane);
#pragma unroll
    for (int i = 0; i < 4; i++)
#pragma unroll
      for (int j = 0; j < 4; j++)
        acc[i][j] = wmma_f16(af[i], bf[j], acc[i][j]);
  }

  const int half = lane >> 4;
#pragma unroll
  for (int i = 0; i < 4; i++) {
#pragma unroll
    for (int j = 0; j < 4; j++) {
      const int col = n0 + 16 * j + (lane & 15);
      const int sec = col >> 10;            // 0=q, 1=k, 2=v
      const int idx = col & 1023;
      const int hh  = idx >> 6, d = idx & 63;
#pragma unroll
      for (int r = 0; r < 8; r++) {
        const int gm = m0 + 16 * i + r + 8 * half;   // global row in [0,4096)
        const int bb = gm >> 11, n = gm & (kN - 1);
        const size_t bh = (size_t)(bb * kHeads + hh);
        const float v = acc[i][j][r];
        if (sec == 0)      qb[(bh * kN + n) * kDh + d] = (_Float16)(v * kQScale);
        else if (sec == 1) kb[(bh * kN + n) * kDh + d] = (_Float16)v;
        else               vt[(bh * kDh + d) * kN + n] = (_Float16)v;
      }
    }
  }
}

// ---------------------------------------------------------------------------
// Flash attention: 1 wave per 32-query tile (two 16-row sub-tiles sharing the
// same K / V^T fragments). Computes S^T = K*Q^T so exp(S) converts
// in-register into the A-fragment of P for the P*V WMMA.
__global__ __launch_bounds__(32)
void attn_kernel(const _Float16* __restrict__ Q,    // [B*H, N, 64] (scaled)
                 const _Float16* __restrict__ Km,   // [B*H, N, 64]
                 const _Float16* __restrict__ VT,   // [B*H, 64, N]
                 _Float16* __restrict__ O) {        // [B, N, H*64]
  const int lane = threadIdx.x;
  const int half = lane >> 4;
  const int q0   = blockIdx.x * 32;
  const int bh   = blockIdx.y;
  const int b    = bh / kHeads, h = bh % kHeads;

  const _Float16* Qb = Q  + (size_t)bh * kN * kDh;
  const _Float16* Kb = Km + (size_t)bh * kN * kDh;
  const _Float16* Vb = VT + (size_t)bh * kDh * kN;

  // Q B-fragments: [q-subtile t][d-chunk c]
  v16h qf[2][2];
#pragma unroll
  for (int t = 0; t < 2; t++)
#pragma unroll
    for (int c = 0; c < 2; c++)
      qf[t][c] = load_b_frag(Qb, kDh, q0 + 16 * t, 32 * c, lane);

  v8f o[2][4] = {};
  float mrun[2] = {-3.0e38f, -3.0e38f};
  float lrun[2] = {0.0f, 0.0f};

  for (int kc = 0; kc < kN; kc += 32) {
    // Prefetch next K/V chunk (address wraps; hint only)
    const int kn = (kc + 32) & (kN - 1);
    __builtin_prefetch(Kb + (size_t)(kn + lane) * kDh, 0, 1);
    __builtin_prefetch(Vb + (size_t)lane * 2 * kN + kn, 0, 1);

    // Shared K fragments for this chunk: [key subtile s][d-chunk c]
    v16h kf[2][2];
#pragma unroll
    for (int s = 0; s < 2; s++)
#pragma unroll
      for (int c = 0; c < 2; c++)
        kf[s][c] = load_a_frag(Kb, kDh, kc + 16 * s, 32 * c, lane);

    // S^T tiles: rows = keys, cols = queries   [key subtile s][q subtile t]
    v8f sacc[2][2] = {};
#pragma unroll
    for (int s = 0; s < 2; s++)
#pragma unroll
      for (int t = 0; t < 2; t++) {
        sacc[s][t] = wmma_f16(kf[s][0], qf[t][0], sacc[s][t]);
        sacc[s][t] = wmma_f16(kf[s][1], qf[t][1], sacc[s][t]);
      }

    // Shared V^T fragments: output d columns  [d tile]
    v16h vf[4];
#pragma unroll
    for (int d = 0; d < 4; d++) vf[d] = load_b_frag(Vb, kN, 16 * d, kc, lane);

    // Per q-subtile online softmax + P*V accumulation
#pragma unroll
    for (int t = 0; t < 2; t++) {
      float mc = sacc[0][t][0];
#pragma unroll
      for (int e = 0; e < 8; e++)
        mc = fmaxf(mc, fmaxf(sacc[0][t][e], sacc[1][t][e]));
      mc = fmaxf(mc, __shfl_xor(mc, 16, 32));
      const float mnew  = fmaxf(mrun[t], mc);
      const float alpha = exp2f(mrun[t] - mnew);
      mrun[t] = mnew;

      // P = exp2(S - m), emitted directly in A-fragment order
      v16h pf;
      float ls = 0.0f;
#pragma unroll
      for (int e = 0; e < 8; e++) {
        const float p0 = exp2f(sacc[0][t][e] - mnew);
        const float p1 = exp2f(sacc[1][t][e] - mnew);
        ls += p0 + p1;
        pf[e]     = (_Float16)p0;   // keys  8h.. 8h+7   of chunk
        pf[8 + e] = (_Float16)p1;   // keys 16+8h..16+8h+7
      }
      ls += __shfl_xor(ls, 16, 32);
      lrun[t] = lrun[t] * alpha + ls;

      // rescale O rows (row m = r + 8*half is query q0+16t+m)
#pragma unroll
      for (int r = 0; r < 8; r++) {
        const float aj = __shfl(alpha, r + 8 * half, 32);
#pragma unroll
        for (int d = 0; d < 4; d++) o[t][d][r] *= aj;
      }

      // O += P * V
#pragma unroll
      for (int d = 0; d < 4; d++) o[t][d] = wmma_f16(pf, vf[d], o[t][d]);
    }
  }

  // normalize and store merged-head f16 activation [B, N, H*64]
  _Float16* ob = O + (size_t)b * kN * kDim + h * kDh + (lane & 15);
#pragma unroll
  for (int t = 0; t < 2; t++) {
#pragma unroll
    for (int r = 0; r < 8; r++) {
      const float inv = 1.0f / __shfl(lrun[t], r + 8 * half, 32);
      const size_t row = (size_t)(q0 + 16 * t + r + 8 * half) * kDim;
#pragma unroll
      for (int d = 0; d < 4; d++)
        ob[row + 16 * d] = (_Float16)(o[t][d][r] * inv);
    }
  }
}

// ---------------------------------------------------------------------------
// Output projection: [4096,1024] x [1024,1024] + bias -> fp32 out
// 8 waves/block, each wave computes a 64x64 tile.
__global__ __launch_bounds__(256)
void proj_gemm(const _Float16* __restrict__ A,    // attn [4096, 1024]
               const _Float16* __restrict__ Bt,   // W_outT [1024, 1024]
               const float* __restrict__ bias,    // [1024]
               float* __restrict__ out) {         // [4096, 1024]
  const int lane = threadIdx.x & 31, w = threadIdx.x >> 5;
  const int m0 = blockIdx.x * 256 + (w >> 1) * 64;
  const int n0 = blockIdx.y * 128 + (w & 1) * 64;

  v8f acc[4][4] = {};
  for (int k = 0; k < kDim; k += 32) {
    v16h af[4], bf[4];
#pragma unroll
    for (int i = 0; i < 4; i++) af[i] = load_a_frag(A,  kDim, m0 + 16 * i, k, lane);
#pragma unroll
    for (int j = 0; j < 4; j++) bf[j] = load_b_frag(Bt, kDim, n0 + 16 * j, k, lane);
#pragma unroll
    for (int i = 0; i < 4; i++)
#pragma unroll
      for (int j = 0; j < 4; j++)
        acc[i][j] = wmma_f16(af[i], bf[j], acc[i][j]);
  }

  const int half = lane >> 4;
#pragma unroll
  for (int i = 0; i < 4; i++) {
#pragma unroll
    for (int j = 0; j < 4; j++) {
      const int col = n0 + 16 * j + (lane & 15);
      const float bv = bias[col];
#pragma unroll
      for (int r = 0; r < 8; r++) {
        const int gm = m0 + 16 * i + r + 8 * half;
        out[(size_t)gm * kDim + col] = acc[i][j][r] + bv;
      }
    }
  }
}

// ---------------------------------------------------------------------------
extern "C" void kernel_launch(void* const* d_in, const int* in_sizes, int n_in,
                              void* d_out, int out_size, void* d_ws,
                              size_t ws_size, hipStream_t stream) {
  (void)in_sizes; (void)n_in; (void)out_size; (void)ws_size;

  const float* x1  = (const float*)d_in[0];
  const float* x2  = (const float*)d_in[1];
  const float* Wq1 = (const float*)d_in[2];
  const float* Wq2 = (const float*)d_in[3];
  const float* Wo1 = (const float*)d_in[4];
  const float* bo1 = (const float*)d_in[5];
  const float* Wo2 = (const float*)d_in[6];
  const float* bo2 = (const float*)d_in[7];
  float* out1 = (float*)d_out;
  float* out2 = out1 + (size_t)kM * kDim;

  // workspace carve-up (f16 elements), total ~96 MiB
  const size_t SZ_X  = (size_t)kM * kDim;               // 4 Mi
  const size_t SZ_WQ = (size_t)kDim * kNcQkv;           // 3 Mi
  const size_t SZ_WO = (size_t)kDim * kDim;             // 1 Mi
  const size_t SZ_HD = (size_t)kB * kHeads * kN * kDh;  // 4 Mi

  _Float16* p = (_Float16*)d_ws;
  _Float16* x1h  = p; p += SZ_X;
  _Float16* x2h  = p; p += SZ_X;
  _Float16* wq1t = p; p += SZ_WQ;
  _Float16* wq2t = p; p += SZ_WQ;
  _Float16* wo1t = p; p += SZ_WO;
  _Float16* wo2t = p; p += SZ_WO;
  _Float16* q1   = p; p += SZ_HD;
  _Float16* k1   = p; p += SZ_HD;
  _Float16* v1t  = p; p += SZ_HD;
  _Float16* q2   = p; p += SZ_HD;
  _Float16* k2   = p; p += SZ_HD;
  _Float16* v2t  = p; p += SZ_HD;
  _Float16* a1   = p; p += SZ_HD;
  _Float16* a2   = p; p += SZ_HD;

  // 1) precision conversion
  const int nx = kM * kDim;
  cvt_f16<<<(nx + 255) / 256, 256, 0, stream>>>(x1, x1h, nx);
  cvt_f16<<<(nx + 255) / 256, 256, 0, stream>>>(x2, x2h, nx);
  const int nwq = kDim * kNcQkv;
  transpose_cvt<<<(nwq + 255) / 256, 256, 0, stream>>>(Wq1, wq1t, kDim, kNcQkv);
  transpose_cvt<<<(nwq + 255) / 256, 256, 0, stream>>>(Wq2, wq2t, kDim, kNcQkv);
  const int nwo = kDim * kDim;
  transpose_cvt<<<(nwo + 255) / 256, 256, 0, stream>>>(Wo1, wo1t, kDim, kDim);
  transpose_cvt<<<(nwo + 255) / 256, 256, 0, stream>>>(Wo2, wo2t, kDim, kDim);

  // 2) QKV projections (scatter to q / k / v^T head layouts)
  dim3 gq(kM / 256, kNcQkv / 128);
  qkv_gemm<<<gq, 256, 0, stream>>>(x1h, wq1t, q1, k1, v1t);
  qkv_gemm<<<gq, 256, 0, stream>>>(x2h, wq2t, q2, k2, v2t);

  // 3) cross attention: dir1 Q=q2 vs K1/V1 ; dir2 Q=q1 vs K2/V2
  dim3 ga(kN / 32, kB * kHeads);
  attn_kernel<<<ga, 32, 0, stream>>>(q2, k1, v1t, a1);
  attn_kernel<<<ga, 32, 0, stream>>>(q1, k2, v2t, a2);

  // 4) output projections (+bias), fp32 result
  dim3 gp(kM / 256, kDim / 128);
  proj_gemm<<<gp, 256, 0, stream>>>(a1, wo1t, bo1, out1);
  proj_gemm<<<gp, 256, 0, stream>>>(a2, wo2t, bo2, out2);
}
